// Sparse_13795434954806
// MI455X (gfx1250) — matrix operand order repaired and checked by reading, not compiled, not run on hardware
//
#include <hip/hip_runtime.h>
#include <hip/hip_bf16.h>
#include <stdint.h>

#define OUT_DIM 4096
#define IN_DIM  4096
#define BATCH   8192

typedef __bf16 bf16_t;
typedef bf16_t v16bf __attribute__((ext_vector_type(16)));
typedef float  v8f   __attribute__((ext_vector_type(8)));
typedef unsigned int v4u __attribute__((ext_vector_type(4)));

union Frag {
    v4u   ui[2];   // 2 x 16 bytes
    v16bf h;       // 16 bf16 x 2 bytes = 32 bytes
};

// ---------------------------------------------------------------------------
// Stage 1: scatter-add COO values into dense W (f32), row-major [out][in]
// ---------------------------------------------------------------------------
__global__ void scatter_coo_kernel(const int* __restrict__ idx,
                                   const float* __restrict__ vals,
                                   float* __restrict__ W, int nnz) {
    int i = blockIdx.x * blockDim.x + threadIdx.x;
    if (i < nnz) {
        int r = idx[2 * i];
        int c = idx[2 * i + 1];
        atomicAdd(&W[(size_t)r * IN_DIM + c], vals[i]);
    }
}

// ---------------------------------------------------------------------------
// Stage 2: f32 -> bf16 (round to nearest even), raw u16 storage
// ---------------------------------------------------------------------------
__global__ void f32_to_bf16_kernel(const float* __restrict__ src,
                                   unsigned short* __restrict__ dst, int n) {
    int i = blockIdx.x * blockDim.x + threadIdx.x;
    if (i < n) {
        unsigned u = __float_as_uint(src[i]);
        unsigned r = (u + 0x7FFFu + ((u >> 16) & 1u)) >> 16;
        dst[i] = (unsigned short)r;
    }
}

// ---------------------------------------------------------------------------
// Stage 3: bf16 WMMA GEMM, out[b,o] = relu(sum_k X[b,k] * W[o,k])
// Block tile 128(M) x 256(N), K-step 32, 8 waves of 32, wave tile 64x64
// (4x4 WMMA tiles, 16 v_wmma per K-step per wave).
// Double-buffered LDS filled with global_load_async_to_lds_b128.
// ---------------------------------------------------------------------------
#define BM 128
#define BN 256
#define BK 32
#define NK (IN_DIM / BK)   // 128 K-iterations

__device__ __forceinline__ void issue_stage(
        const unsigned short* __restrict__ Xb,
        const unsigned short* __restrict__ Wb,
        unsigned short (*sA)[BK], unsigned short (*sB)[BK],
        int m_base, int n_base, int k0, int tid) {
    // A tile: 128 rows x 32 bf16 = 8 KB = 512 x 16B chunks -> 2 per thread
#pragma unroll
    for (int j = 0; j < 2; ++j) {
        int c   = tid + j * 256;
        int row = c >> 2;        // 0..127
        int q   = c & 3;         // 16-byte chunk within 64-byte row
        const unsigned short* ga = Xb + (size_t)(m_base + row) * IN_DIM + k0 + q * 8;
        unsigned aoff = (unsigned)(uintptr_t)&sA[row][q * 8];
        asm volatile("global_load_async_to_lds_b128 %0, %1, off"
                     :: "v"(aoff), "v"(ga) : "memory");
    }
    // B tile: 256 rows x 32 bf16 = 16 KB = 1024 x 16B chunks -> 4 per thread
#pragma unroll
    for (int j = 0; j < 4; ++j) {
        int c   = tid + j * 256;
        int row = c >> 2;        // 0..255
        int q   = c & 3;
        const unsigned short* gb = Wb + (size_t)(n_base + row) * IN_DIM + k0 + q * 8;
        unsigned boff = (unsigned)(uintptr_t)&sB[row][q * 8];
        asm volatile("global_load_async_to_lds_b128 %0, %1, off"
                     :: "v"(boff), "v"(gb) : "memory");
    }
}

__global__ __launch_bounds__(256) void gemm_bf16_wmma_relu_kernel(
        const unsigned short* __restrict__ Xb,   // [BATCH][IN_DIM] bf16
        const unsigned short* __restrict__ Wb,   // [OUT_DIM][IN_DIM] bf16
        float* __restrict__ out) {               // [BATCH][OUT_DIM] f32
    __shared__ __align__(16) unsigned short sA[2][BM][BK];   // 2 x 8 KB
    __shared__ __align__(16) unsigned short sB[2][BN][BK];   // 2 x 16 KB

    const int tid    = threadIdx.x;
    const int m_base = blockIdx.y * BM;
    const int n_base = blockIdx.x * BN;

    const int lane = tid & 31;
    const int lr   = lane & 15;      // lane % 16
    const int hi   = lane >> 4;      // upper half of wave
    const int wave = tid >> 5;       // 0..7
    const int wm   = wave & 1;       // 2 waves along M (64 rows each)
    const int wn   = wave >> 1;      // 4 waves along N (64 cols each)

    v8f acc[4][4];
#pragma unroll
    for (int t = 0; t < 4; ++t)
#pragma unroll
        for (int n2 = 0; n2 < 4; ++n2)
            acc[t][n2] = (v8f){0.f, 0.f, 0.f, 0.f, 0.f, 0.f, 0.f, 0.f};

    // prologue: stage 0
    issue_stage(Xb, Wb, sA[0], sB[0], m_base, n_base, 0, tid);

    for (int s = 0; s < NK; ++s) {
        const int buf = s & 1;
        if (s + 1 < NK) {
            issue_stage(Xb, Wb, sA[buf ^ 1], sB[buf ^ 1],
                        m_base, n_base, (s + 1) * BK, tid);
            // drain stage s (6 oldest asyncs); stage s+1's 6 stay in flight
            asm volatile("s_wait_asynccnt 6" ::: "memory");
        } else {
            asm volatile("s_wait_asynccnt 0" ::: "memory");
        }
        __syncthreads();   // all waves' LDS fills for stage s visible

        // --- A fragments: 16x32 bf16, ISA layout ---
        // lanes 0-15 : row M=lr, K 0-7 in v0-3, K 16-23 in v4-7
        // lanes 16-31: row M=lr, K 8-15 in v0-3, K 24-31 in v4-7
        Frag a[4];
#pragma unroll
        for (int t = 0; t < 4; ++t) {
            const unsigned short* p = &sA[buf][wm * 64 + t * 16 + lr][hi * 8];
            a[t].ui[0] = *(const v4u*)p;
            a[t].ui[1] = *(const v4u*)(p + 16);   // +32 bytes
        }
        // --- B fragments: 32x16 bf16 (K x N), W row o is column n=o ---
        // lanes 0-15 : col N=lr, K 0-7, 16-23 ; lanes 16-31: col N=lr, K 8-15, 24-31
        Frag b[4];
#pragma unroll
        for (int n2 = 0; n2 < 4; ++n2) {
            const unsigned short* p = &sB[buf][wn * 64 + n2 * 16 + lr][hi * 8];
            b[n2].ui[0] = *(const v4u*)p;
            b[n2].ui[1] = *(const v4u*)(p + 16);   // +32 bytes
        }

#pragma unroll
        for (int t = 0; t < 4; ++t)
#pragma unroll
            for (int n2 = 0; n2 < 4; ++n2)
                acc[t][n2] = __builtin_amdgcn_wmma_f32_16x16x32_bf16(
                        false, a[t].h, false, b[n2].h,
                        (short)0, acc[t][n2], false, false);

        __syncthreads();   // buf may be overwritten by stage s+2 next iter
    }

    // --- store with fused ReLU; C layout: VGPR r -> M = r + 8*hi, N = lr ---
#pragma unroll
    for (int t = 0; t < 4; ++t) {
#pragma unroll
        for (int n2 = 0; n2 < 4; ++n2) {
            const int row0 = m_base + wm * 64 + t * 16 + hi * 8;
            const int col  = n_base + wn * 64 + n2 * 16 + lr;
#pragma unroll
            for (int r = 0; r < 8; ++r) {
                out[(size_t)(row0 + r) * OUT_DIM + col] =
                    fmaxf(acc[t][n2][r], 0.0f);
            }
        }
    }
}

// ---------------------------------------------------------------------------
// Host launcher
// ---------------------------------------------------------------------------
extern "C" void kernel_launch(void* const* d_in, const int* in_sizes, int n_in,
                              void* d_out, int out_size, void* d_ws, size_t ws_size,
                              hipStream_t stream) {
    const float* x    = (const float*)d_in[0];
    const int*   idx  = (const int*)d_in[1];   // (N_NNZ, 2) flat
    const float* vals = (const float*)d_in[2];
    float*       out  = (float*)d_out;
    const int nnz = in_sizes[2];

    // workspace layout: Wf32 (64 MB) | Wbf16 (32 MB) | Xbf16 (64 MB)
    const size_t w_elems = (size_t)OUT_DIM * IN_DIM;
    const size_t x_elems = (size_t)BATCH * IN_DIM;
    float*          Wf = (float*)d_ws;
    unsigned short* Wb = (unsigned short*)((char*)d_ws + w_elems * sizeof(float));
    unsigned short* Xb = (unsigned short*)((char*)d_ws + w_elems * sizeof(float)
                                           + w_elems * sizeof(unsigned short));

    // 1) zero dense W
    hipMemsetAsync(Wf, 0, w_elems * sizeof(float), stream);
    // 2) scatter-add nnz
    scatter_coo_kernel<<<(nnz + 255) / 256, 256, 0, stream>>>(idx, vals, Wf, nnz);
    // 3) convert W and X to bf16
    f32_to_bf16_kernel<<<((int)w_elems + 255) / 256, 256, 0, stream>>>(Wf, Wb, (int)w_elems);
    f32_to_bf16_kernel<<<((int)x_elems + 255) / 256, 256, 0, stream>>>(x, Xb, (int)x_elems);
    // 4) WMMA GEMM + ReLU: grid = (N/BN, M/BM)
    gemm_bf16_wmma_relu_kernel<<<dim3(OUT_DIM / BN, BATCH / BM), 256, 0, stream>>>(
        Xb, Wb, out);
}
